// GCNLayer_56341380989220
// MI455X (gfx1250) — matrix operand order repaired and checked by reading, not compiled, not run on hardware
//
#include <hip/hip_runtime.h>
#include <hip/hip_bf16.h>

// ---------------------------------------------------------------------------
// GCN layer for MI455X (gfx1250):
//   h   = x @ W^T              (bf16 WMMA, f32 accumulate, h stored bf16)
//   out = segment_sum(vals * h[col], row)   (f32 hardware atomics in L2)
//
// v3: two LDS base pointers keep every B-fragment ds_load offset inside the
// 16-bit immediate (no in-loop address VALU -> no WMMA->VALU hazard NOPs),
// and B fragments are software-pipelined (load b[nt+1] while WMMA consumes
// b[nt]) so ds_loads overlap the matrix pipe instead of draining dscnt to 0.
// ---------------------------------------------------------------------------

typedef __attribute__((ext_vector_type(16))) __bf16 v16bf;
typedef __attribute__((ext_vector_type(8)))  __bf16 v8bf;
typedef __attribute__((ext_vector_type(8)))  float  v8f;

#define DIM 256
// LDS row stride (bf16 elements): 272*2 = 544 B -> rows stay 32B-aligned
// (544 = 17*32) and the 16 per-fragment row readers spread across 8 banks.
#define WSTRIDE 272

// --------------------------------------------------------------------------
// Kernel 1: zero the output (we accumulate into it with atomics).
// --------------------------------------------------------------------------
__global__ void __launch_bounds__(256) zero_f4_kernel(float4* __restrict__ p, size_t n4) {
    size_t i = (size_t)blockIdx.x * blockDim.x + threadIdx.x;
    if (i < n4) p[i] = make_float4(0.f, 0.f, 0.f, 0.f);
}

// --------------------------------------------------------------------------
// Kernel 2: one-time W (f32) -> Wb (bf16) conversion. 256x256 = 64K elems.
// --------------------------------------------------------------------------
__global__ void __launch_bounds__(256) wconv_kernel(const float* __restrict__ W,
                                                    __bf16* __restrict__ Wb, int n) {
    int i = blockIdx.x * blockDim.x + threadIdx.x;
    if (i < n) Wb[i] = (__bf16)W[i];
}

// --------------------------------------------------------------------------
// Kernel 3: h = x @ W^T via v_wmma_f32_16x16x32_bf16, W staged in LDS.
// One wave per 16-row M-tile. 16 accumulators (16x16 f32) cover all 256
// output columns; K-loop is 8 steps of 32.
//
// A fragment (16x32 bf16, contiguous K): lane r (0-15) holds row M=r,
//   K = 0..15; lane r+16 holds row M=r, K = 16..31 (same order as B).
// B fragment (32x16 bf16): B[k][n] = W[n][k]. lane r (0-15) holds column
//   N=r, K=0..15; lane r+16 holds column N=r, K=16..31 -> 16 consecutive
//   bf16 from LDS row n => two aligned ds_load_b128 with immediate offsets.
// C/D (16x16 f32): elem i of lane r(+16*half) is row (half*8+i), col r.
// --------------------------------------------------------------------------
__global__ void __launch_bounds__(256) gemm_wmma_kernel(const float*  __restrict__ x,
                                                        const __bf16* __restrict__ Wb,
                                                        __bf16*       __restrict__ h,
                                                        int nRows) {
    __shared__ __attribute__((aligned(32))) __bf16 Wlds[DIM * WSTRIDE];

    // ---- cooperative, coalesced fill of W into LDS (8192 x 16B chunks) ----
    {
        const int tid = threadIdx.x;
#pragma unroll
        for (int it = 0; it < 32; ++it) {
            const int chunk = it * 256 + tid;     // 0..8191
            const int e     = chunk * 8;          // element index (mult of 8)
            const int row   = e >> 8;             // 8 consecutive bf16 stay in-row
            const int col   = e & 255;
            *(v8bf*)&Wlds[row * WSTRIDE + col] = *(const v8bf*)(Wb + e);
        }
    }
    __syncthreads();

    const int wave = blockIdx.x * (blockDim.x >> 5) + (threadIdx.x >> 5);
    const int lane = threadIdx.x & 31;
    const int m0   = wave * 16;
    if (m0 >= nRows) return;

    const int half = lane >> 4;   // 0 or 1
    const int r    = lane & 15;

    // Two LDS base pointers: n-tiles 0..7 start at row r, n-tiles 8..15 at
    // row 128+r. Per-fragment byte offset = (nt&7)*16*WSTRIDE*2 + k0*2
    // <= 60928 + 448, inside the 16-bit ds immediate -> no in-loop VALU.
    const __bf16* wbase0 = &Wlds[(unsigned)r * WSTRIDE + half * 16];
    const __bf16* wbase1 = &Wlds[(unsigned)(128 + r) * WSTRIDE + half * 16];

    v8f acc[16] = {};

    const float* arow = x + (size_t)(m0 + r) * DIM;

#pragma unroll
    for (int k0 = 0; k0 < DIM; k0 += 32) {
        // ---- A fragment: two 32B f32 chunks -> 16 bf16 ----
        const float4* a_lo = (const float4*)(arow + k0 + half * 16);      // K = half*16 .. +7
        const float4* a_hi = (const float4*)(arow + k0 + half * 16 + 8);  // K = half*16+8 .. +7
        float4 p0 = a_lo[0], p1 = a_lo[1];
        float4 p2 = a_hi[0], p3 = a_hi[1];
        v16bf a;
        a[0]  = (__bf16)p0.x; a[1]  = (__bf16)p0.y; a[2]  = (__bf16)p0.z; a[3]  = (__bf16)p0.w;
        a[4]  = (__bf16)p1.x; a[5]  = (__bf16)p1.y; a[6]  = (__bf16)p1.z; a[7]  = (__bf16)p1.w;
        a[8]  = (__bf16)p2.x; a[9]  = (__bf16)p2.y; a[10] = (__bf16)p2.z; a[11] = (__bf16)p2.w;
        a[12] = (__bf16)p3.x; a[13] = (__bf16)p3.y; a[14] = (__bf16)p3.z; a[15] = (__bf16)p3.w;

        // ---- 16 n-tiles, B software-pipelined one fragment ahead ----
        v16bf bcur = *(const v16bf*)(wbase0 + k0);
#pragma unroll
        for (int nt = 0; nt < 16; ++nt) {
            v16bf bnext;
            if (nt < 15) {
                const int ntn = nt + 1;
                const __bf16* base = (ntn < 8) ? wbase0 : wbase1;
                bnext = *(const v16bf*)(base + (ntn & 7) * 16 * WSTRIDE + k0);
            }
            acc[nt] = __builtin_amdgcn_wmma_f32_16x16x32_bf16(
                /*neg_a=*/false, a, /*neg_b=*/false, bcur,
                /*c_mod=*/(short)0, acc[nt], /*reuse_a=*/false, /*reuse_b=*/false);
            bcur = bnext;
        }
    }

    // ---- store 16x256 tile as bf16 ----
#pragma unroll
    for (int nt = 0; nt < 16; ++nt) {
#pragma unroll
        for (int i = 0; i < 8; ++i) {
            h[(size_t)(m0 + half * 8 + i) * DIM + nt * 16 + r] = (__bf16)acc[nt][i];
        }
    }
}

// --------------------------------------------------------------------------
// Kernel 4: sparse COO aggregation. One wave (32 lanes) per edge; each lane
// handles 8 consecutive columns: one 16B bf16 vector load of h[col], scale
// by edge value, 8 hardware f32 atomic adds into out[row] (L2-resident).
// --------------------------------------------------------------------------
__global__ void __launch_bounds__(256) scatter_kernel(const __bf16* __restrict__ h,
                                                      const float*  __restrict__ vals,
                                                      const int*    __restrict__ rows,
                                                      const int*    __restrict__ cols,
                                                      float*        __restrict__ out,
                                                      int nEdges) {
    const int wave = blockIdx.x * (blockDim.x >> 5) + (threadIdx.x >> 5);
    const int lane = threadIdx.x & 31;
    if (wave >= nEdges) return;

    const int   c  = cols[wave];
    const int   rr = rows[wave];
    const float v  = vals[wave];

    const v8bf hv = *(const v8bf*)(h + (size_t)c * DIM + lane * 8);
    float* orow = out + (size_t)rr * DIM + lane * 8;

#pragma unroll
    for (int j = 0; j < 8; ++j) {
        unsafeAtomicAdd(orow + j, v * (float)hv[j]);   // global_atomic_add_f32
    }
}

// --------------------------------------------------------------------------
// Launch: zero(out) -> W->bf16 -> WMMA GEMM -> edge scatter
// --------------------------------------------------------------------------
extern "C" void kernel_launch(void* const* d_in, const int* in_sizes, int n_in,
                              void* d_out, int out_size, void* d_ws, size_t ws_size,
                              hipStream_t stream) {
    const float* x   = (const float*)d_in[0];
    const float* W   = (const float*)d_in[1];
    const float* ev  = (const float*)d_in[2];
    const int*   er  = (const int*)d_in[3];
    const int*   ec  = (const int*)d_in[4];
    float*       out = (float*)d_out;

    const int nNodes = in_sizes[0] / DIM;   // 100000
    const int wElems = in_sizes[1];         // 65536
    const int nEdges = in_sizes[2];         // 3200000

    // workspace layout: [ h bf16 : nNodes*256 ][ Wb bf16 : 256*256 ]
    __bf16* h  = (__bf16*)d_ws;
    __bf16* Wb = h + (size_t)nNodes * DIM;

    // 1) zero output (accumulated via atomics)
    const size_t n4 = (size_t)out_size / 4;
    zero_f4_kernel<<<(unsigned)((n4 + 255) / 256), 256, 0, stream>>>((float4*)d_out, n4);

    // 2) W -> bf16
    wconv_kernel<<<(wElems + 255) / 256, 256, 0, stream>>>(W, Wb, wElems);

    // 3) h = x @ W^T (bf16 WMMA, f32 accumulate, W in LDS)
    const int mTiles  = (nNodes + 15) / 16;         // 6250
    const int gBlocks = (mTiles + 7) / 8;           // 8 waves per 256-thread block
    gemm_wmma_kernel<<<gBlocks, 256, 0, stream>>>(x, Wb, h, nNodes);

    // 4) edge scatter: one wave per edge
    const int sBlocks = (nEdges + 7) / 8;
    scatter_kernel<<<sBlocks, 256, 0, stream>>>(h, ev, er, ec, out, nEdges);
}
